// ImplicitKernelAttention_82592221102311
// MI455X (gfx1250) — matrix operand order repaired and checked by reading, not compiled, not run on hardware
//
#include <hip/hip_runtime.h>
#include <hip/hip_bf16.h>
#include <math.h>
#include <stdint.h>

typedef __attribute__((ext_vector_type(16))) _Float16 v16h;
typedef __attribute__((ext_vector_type(8)))  float    v8f;

#define BN 8
#define SN 2048
#define DN 256
#define DVN 256
#define RN 128
#define TRN 256          // 2R
#define SEP (SN + 8)     // padded LDS row stride for f32 energy (floats)
#define EPSC 1e-5f
#define INV2SQRTD (1.0f / 32.0f)   // 1/(2*sqrt(256))

// LDS layout offsets (bytes) for the attention kernel
#define SMEM_PHIQ   0
#define SMEM_E      (16 * TRN * 2)                          // 8192
#define SMEM_AH     (SMEM_E + 16 * SEP * 4)                 // + 131584
#define SMEM_QN     (SMEM_AH + 16 * SN * 2)                 // + 65536
#define SMEM_TOTAL  (SMEM_QN + 64)

// ---------------------------------------------------------------------------
// Kernel 1: random-feature projection. grid=(B*S, 2), block=128.
//   y==0: query -> phiQ/qn ; y==1: keys -> phiK/kn
// ---------------------------------------------------------------------------
__global__ void ika_proj_kernel(const float* __restrict__ query,
                                const float* __restrict__ keys,
                                const float* __restrict__ W_w,
                                const float* __restrict__ W_b,
                                _Float16* __restrict__ phiQ,
                                _Float16* __restrict__ phiK,
                                float* __restrict__ qn,
                                float* __restrict__ kn) {
    const int row = blockIdx.x;                 // 0 .. B*S-1
    const int isK = blockIdx.y;
    const int r   = threadIdx.x;                // 0 .. 127 (feature)

    const float* __restrict__ x = (isK ? keys : query) + (size_t)row * DN;
    _Float16* __restrict__ phi  = (isK ? phiK : phiQ) + (size_t)row * TRN;
    float* __restrict__ xn      = isK ? kn : qn;

    __shared__ float xrow[DN];
    for (int d = r; d < DN; d += 128) xrow[d] = x[d];
    __syncthreads();

    float acc = W_b[r];
    const float* __restrict__ w = W_w + (size_t)r * DN;
#pragma unroll 4
    for (int d = 0; d < DN; ++d) acc = fmaf(xrow[d], w[d], acc);

    if (r < 32) {
        float s = 0.f;
        for (int d = r; d < DN; d += 32) { float v = xrow[d]; s = fmaf(v, v, s); }
#pragma unroll
        for (int off = 16; off; off >>= 1) s += __shfl_xor(s, off, 32);
        if (r == 0) xn[row] = s;
    }

    phi[r]      = (_Float16)__cosf(acc);
    phi[RN + r] = (_Float16)__sinf(acc);
}

// ---------------------------------------------------------------------------
// Kernel 2: tiled transpose vals[b][k][n] (f32) -> valsT[b][n][k] (f16)
// grid=(B, S/32, DV/32), block=256. Makes AV B-fragments contiguous 32B reads.
// ---------------------------------------------------------------------------
__global__ void ika_vt_kernel(const float* __restrict__ vals,
                              _Float16* __restrict__ valsT) {
    __shared__ _Float16 t[32][33];
    const int b  = blockIdx.x;
    const int k0 = blockIdx.y * 32;
    const int n0 = blockIdx.z * 32;
    const int tx = threadIdx.x & 31;
    const int ty = threadIdx.x >> 5;            // 0..7

    const float* __restrict__ src = vals + ((size_t)b * SN + k0) * DVN + n0;
#pragma unroll
    for (int r = 0; r < 4; ++r) {
        const int kk = ty + r * 8;
        t[kk][tx] = (_Float16)src[(size_t)kk * DVN + tx];
    }
    __syncthreads();
    _Float16* __restrict__ dst = valsT + ((size_t)b * DVN + n0) * SN + k0;
#pragma unroll
    for (int r = 0; r < 4; ++r) {
        const int nn = ty + r * 8;
        dst[(size_t)nn * SN + tx] = t[tx][nn];
    }
}

// ---------------------------------------------------------------------------
// Kernel 3: energy (WMMA) -> mask -> softmax -> attn@V (WMMA).
// grid = B*(S/16) blocks, 256 threads (8 wave32), dynamic LDS (~205 KB).
// ---------------------------------------------------------------------------
__global__ void ika_attn_kernel(const _Float16* __restrict__ phiQ,
                                const _Float16* __restrict__ phiK,
                                const float* __restrict__ qn,
                                const float* __restrict__ kn,
                                const _Float16* __restrict__ valsT,
                                float* __restrict__ outp,
                                float* __restrict__ attnp) {
    extern __shared__ char smem[];
    _Float16* sPhiQ = (_Float16*)(smem + SMEM_PHIQ);   // 16 x 256 f16
    float*    sE    = (float*)   (smem + SMEM_E);      // 16 x SEP f32
    _Float16* sAh   = (_Float16*)(smem + SMEM_AH);     // 16 x 2048 f16 (attn)
    float*    sQn   = (float*)   (smem + SMEM_QN);     // 16 f32

    const int tid  = threadIdx.x;
    const int wave = tid >> 5;
    const int lane = tid & 31;
    const int half = lane >> 4;
    const int l16  = lane & 15;

    const int b  = blockIdx.x / (SN / 16);
    const int q0 = (blockIdx.x % (SN / 16)) * 16;

    // ---- async-stage phiQ tile (contiguous 8 KB) via CDNA5 async copy ----
    {
        const uint64_t g = (uint64_t)(phiQ + ((size_t)(b * SN + q0)) * TRN)
                         + (uint64_t)tid * 32;
        const unsigned ldsOff = SMEM_PHIQ + (unsigned)tid * 32;
        asm volatile(
            "global_load_async_to_lds_b128 %0, %1, off\n\t"
            "global_load_async_to_lds_b128 %0, %1, off offset:16"
            :: "v"(ldsOff), "v"(g) : "memory");
    }
    if (tid < 16) sQn[tid] = qn[b * SN + q0 + tid];
    asm volatile("s_wait_asynccnt 0" ::: "memory");
    __syncthreads();

    // ---- energy: each wave owns key-tiles wave, wave+8, ... (16 tiles) ----
    const _Float16* __restrict__ phiKb = phiK + (size_t)b * SN * TRN;
    const _Float16* __restrict__ aBase = sPhiQ + l16 * TRN;
    for (int kt = wave; kt < SN / 16; kt += 8) {
        const int n0 = kt * 16;
        const _Float16* browBase =
            phiKb + (size_t)(n0 + l16) * TRN + half * 16;
        __builtin_prefetch(browBase + (size_t)128 * TRN, 0, 1); // next tile
        v8f acc = {};
#pragma unroll
        for (int kb = 0; kb < 8; ++kb) {
            // A fragment from LDS: 16 contiguous halves per lane (32 bytes)
            const _Float16* ar = aBase + kb * 32;
            v16h a;
#pragma unroll
            for (int j = 0; j < 16; ++j) a[j] = ar[half * 16 + j];
            // B fragment: 16 contiguous halves per lane (32 bytes)
            const _Float16* brow = browBase + kb * 32;
            v16h bf;
#pragma unroll
            for (int j = 0; j < 16; ++j) bf[j] = brow[j];
            acc = __builtin_amdgcn_wmma_f32_16x16x32_f16(
                false, a, false, bf, (short)0, acc, false, false);
        }
        const int   ncol = n0 + l16;
        const float knv  = kn[b * SN + ncol];
#pragma unroll
        for (int i = 0; i < 8; ++i) {
            const int m = half * 8 + i;                  // C layout: M=half*8+i, N=l16
            float e   = acc[i] * (1.0f / 128.0f);
            float pre = __logf(e * e + EPSC) + (sQn[m] + knv) * INV2SQRTD;
            if (ncol > q0 + m) pre = -INFINITY;          // causal mask
            sE[m * SEP + ncol] = pre;
        }
    }
    __syncthreads();

    // ---- softmax: 2 rows per wave; write attn to global + f16 LDS copy ----
#pragma unroll
    for (int rr = 0; rr < 2; ++rr) {
        const int m = wave * 2 + rr;
        float* row = sE + m * SEP;
        float mx = -INFINITY;
        for (int j = lane; j < SN; j += 32) mx = fmaxf(mx, row[j]);
#pragma unroll
        for (int off = 16; off; off >>= 1) mx = fmaxf(mx, __shfl_xor(mx, off, 32));
        float sum = 0.f;
        for (int j = lane; j < SN; j += 32) {
            float v = __expf(row[j] - mx);
            row[j] = v;
            sum += v;
        }
#pragma unroll
        for (int off = 16; off; off >>= 1) sum += __shfl_xor(sum, off, 32);
        const float inv = 1.0f / sum;
        float* __restrict__ arow = attnp + ((size_t)b * SN + q0 + m) * SN;
        _Float16* __restrict__ hrow = sAh + (size_t)m * SN;
        for (int j = lane; j < SN; j += 32) {
            float v = row[j] * inv;
            arow[j] = v;
            hrow[j] = (_Float16)v;
        }
    }
    __syncthreads();

    // ---- out = attn @ V : each wave owns n-tiles wave, wave+8 (K = 2048) ----
    const _Float16* __restrict__ vTb   = valsT + (size_t)b * DVN * SN;
    const _Float16* __restrict__ aAttn = sAh + (size_t)l16 * SN;
    for (int nt = wave; nt < DVN / 16; nt += 8) {
        const int n0 = nt * 16;
        const _Float16* __restrict__ bColBase =
            vTb + (size_t)(n0 + l16) * SN + half * 16;
        v8f acc = {};
        for (int kb = 0; kb < SN / 32; ++kb) {
            const int kbase = kb * 32;
            // A from LDS attn f16: 16 contiguous halves per lane (32 bytes)
            const _Float16* ar = aAttn + kbase;
            v16h a;
#pragma unroll
            for (int j = 0; j < 16; ++j) a[j] = ar[half * 16 + j];
            // B[k][n] = valsT[n][k]: 16 contiguous halves per lane (32 bytes)
            const _Float16* bcol = bColBase + kbase;
            __builtin_prefetch(bcol + 32, 0, 1);         // next K step
            v16h bf;
#pragma unroll
            for (int j = 0; j < 16; ++j) bf[j] = bcol[j];
            acc = __builtin_amdgcn_wmma_f32_16x16x32_f16(
                false, a, false, bf, (short)0, acc, false, false);
        }
#pragma unroll
        for (int i = 0; i < 8; ++i) {
            const int m = half * 8 + i;
            outp[((size_t)b * SN + q0 + m) * DVN + n0 + l16] = acc[i];
        }
    }
}

// ---------------------------------------------------------------------------
extern "C" void kernel_launch(void* const* d_in, const int* in_sizes, int n_in,
                              void* d_out, int out_size, void* d_ws, size_t ws_size,
                              hipStream_t stream) {
    const float* query = (const float*)d_in[0];
    const float* keys  = (const float*)d_in[1];
    const float* vals  = (const float*)d_in[2];
    // d_in[3] = mask (deterministic causal tril; recomputed in-kernel)
    const float* W_w   = (const float*)d_in[4];
    const float* W_b   = (const float*)d_in[5];

    float* outp  = (float*)d_out;                          // [B,S,DV]
    float* attnp = outp + (size_t)BN * SN * DVN;           // [B,S,S]

    // workspace layout
    _Float16* phiQ  = (_Float16*)d_ws;                     // B*S*2R f16
    _Float16* phiK  = phiQ + (size_t)BN * SN * TRN;
    _Float16* valsT = phiK + (size_t)BN * SN * TRN;        // B*DV*S f16
    float*    qn    = (float*)(valsT + (size_t)BN * DVN * SN);
    float*    kn    = qn + (size_t)BN * SN;

    ika_proj_kernel<<<dim3(BN * SN, 2, 1), 128, 0, stream>>>(
        query, keys, W_w, W_b, phiQ, phiK, qn, kn);

    ika_vt_kernel<<<dim3(BN, SN / 32, DVN / 32), 256, 0, stream>>>(vals, valsT);

    ika_attn_kernel<<<dim3(BN * (SN / 16), 1, 1), 256, SMEM_TOTAL, stream>>>(
        phiQ, phiK, qn, kn, valsT, outp, attnp);
}
